// BiMambaLayer_27376121544953
// MI455X (gfx1250) — matrix operand order repaired and checked by reading, not compiled, not run on hardware
//
#include <hip/hip_runtime.h>
#include <hip/hip_bf16.h>

// ---------------------------------------------------------------------------
// BiMamba (fwd+bwd) for gfx1250 / MI455X.
// GEMMs via v_wmma_f32_16x16x32_bf16 (fp32 accumulate); the large in_proj GEMM
// stages 128x128 block tiles through LDS with async global->LDS copies
// (ASYNCcnt-tracked, double buffered); scan in fp32 registers.
// ---------------------------------------------------------------------------

typedef __bf16 bf16_t;
typedef __attribute__((ext_vector_type(16))) __bf16 v16bf;
typedef __attribute__((ext_vector_type(8)))  __bf16 v8bf;
typedef __attribute__((ext_vector_type(8)))  float  v8f;

#define BATCH   8
#define SEQ     1024
#define DMODEL  512
#define DINNER  1024
#define DSTATE  16
#define DTRANK  32
#define TOK     (BATCH * SEQ)      // 8192 tokens

__device__ __forceinline__ v8f wmma_bf16(v16bf a, v16bf b, v8f c) {
    // D = A(16x32) * B(32x16) + C, fp32 accumulate
    return __builtin_amdgcn_wmma_f32_16x16x32_bf16(
        /*neg_a=*/false, a, /*neg_b=*/false, b,
        /*c_mod=*/(short)0, c, /*reuse_a=*/false, /*reuse_b=*/false);
}

// A fragment: row-major bf16 A[M x lda], tile (m0,k0), 16x32.
// Layout: lanes 0-15 hold rows m0..m0+15 with K = k0+{0..15};
//         lanes 16-31 same rows with K = k0+{16..31}.
__device__ __forceinline__ v16bf load_a_frag(const bf16_t* A, int lda,
                                             int m0, int k0, int lane) {
    const int half = (lane >> 4) & 1;
    const int m    = m0 + (lane & 15);
    const bf16_t* p = A + (size_t)m * lda + k0 + half * 16;
    v8bf lo = *(const v8bf*)(p);        // K = k0 + half*16 + [0..7]
    v8bf hi = *(const v8bf*)(p + 8);    // K = k0 + half*16 + 8 + [0..7]
    v16bf r;
#pragma unroll
    for (int i = 0; i < 8; ++i) { r[i] = lo[i]; r[8 + i] = hi[i]; }
    return r;
}

// B fragment for C = A @ W^T with row-major W[N x ldw]: B[k][n] = W[n][k].
// Lane holds column n = n0 + (lane&15); K = k0 + 16*(lane>>4) + [0..15],
// one contiguous 32-byte run of row n of W.
__device__ __forceinline__ v16bf load_bT_frag(const bf16_t* W, int ldw,
                                              int n0, int k0, int lane) {
    const int half = (lane >> 4) & 1;
    const int n    = n0 + (lane & 15);
    return *(const v16bf*)(W + (size_t)n * ldw + k0 + half * 16);
}

// C/D layout: VGPR r -> row m0 + r + 8*(lane>>4); col n0 + (lane&15).
__device__ __forceinline__ void store_c(float* C, int ldc, int m0, int n0,
                                        int lane, v8f acc) {
    const int half = lane >> 4;
    const int n    = n0 + (lane & 15);
#pragma unroll
    for (int r = 0; r < 8; ++r)
        C[(size_t)(m0 + r + 8 * half) * ldc + n] = acc[r];
}

// --------------------------- async global -> LDS ---------------------------
// Generic pointers into LDS carry the LDS byte offset in addr[31:0]
// (aperture layout, ISA 10.2), so truncation yields the wave-relative offset.
__device__ __forceinline__ unsigned int lds_off_of(const void* p) {
    return (unsigned int)(unsigned long long)(uintptr_t)p;
}
// GLOBAL_LOAD_ASYNC_TO_LDS_B128: per-lane LDS[vdst] = MEM[vaddr]; ASYNCcnt++.
__device__ __forceinline__ void async_ld_b128(unsigned int lds_off, const void* g) {
    asm volatile("global_load_async_to_lds_b128 %0, %1, off"
                 :: "v"(lds_off), "v"((unsigned long long)(uintptr_t)g)
                 : "memory");
}
__device__ __forceinline__ void wait_async_le4() {
    asm volatile("s_wait_asynccnt 0x4" ::: "memory");
}
__device__ __forceinline__ void wait_async_0() {
    asm volatile("s_wait_asynccnt 0x0" ::: "memory");
}

// ---------------------------------------------------------------------------
// f32 -> bf16 conversion
// ---------------------------------------------------------------------------
__global__ void k_cvt_bf16(const float* __restrict__ src,
                           bf16_t* __restrict__ dst, int n) {
    int i = blockIdx.x * 256 + threadIdx.x;
    if (i < n) dst[i] = (bf16_t)src[i];
}

// ---------------------------------------------------------------------------
// P[8192,2048] = Xb[8192,512] @ in_w^T   (both directions)
// LDS-staged, double-buffered 128x128 block tile; 8 waves (4m x 2n), each
// wave owns a 32x64 sub-tile (2x4 fragments).  A/B K-panels (128x32 bf16,
// 8 KB each) are copied with async global->LDS b128 (4 copy instructions per
// thread per panel) so the next panel streams in while WMMAs consume the
// current one.  grid(64, 16, 2), block 256.
// ---------------------------------------------------------------------------
__global__ void __launch_bounds__(256)
k_gemm_inproj(const bf16_t* __restrict__ Xb,
              const bf16_t* __restrict__ Wf, const bf16_t* __restrict__ Wb,
              float* __restrict__ Pf, float* __restrict__ Pb) {
    __shared__ bf16_t smem[2 * 8192];   // [buf][A 4096 | B 4096] = 32 KB

    const int tid  = threadIdx.x;
    const int lane = tid & 31;
    const int wave = tid >> 5;
    const int wm   = wave >> 1;          // 0..3 -> 32-row strip
    const int wn   = wave & 1;           // 0..1 -> 64-col strip
    const int mBase = blockIdx.x * 128;
    const int nBase = blockIdx.y * 128;
    const bf16_t* W = blockIdx.z ? Wb : Wf;
    float*        C = blockIdx.z ? Pb : Pf;

    const unsigned int lds0 = lds_off_of(&smem[0]);
    const int q0 = tid, q1 = tid + 256;  // two 16B chunks per thread per panel

    auto issue = [&](int buf, int k0) {
        const unsigned int aoff = lds0 + buf * 16384;
        const unsigned int boff = aoff + 8192;
        {
            const int r = q0 >> 2, kc = (q0 & 3) * 8;
            async_ld_b128(aoff + q0 * 16, Xb + (size_t)(mBase + r) * DMODEL + k0 + kc);
            async_ld_b128(boff + q0 * 16, W  + (size_t)(nBase + r) * DMODEL + k0 + kc);
        }
        {
            const int r = q1 >> 2, kc = (q1 & 3) * 8;
            async_ld_b128(aoff + q1 * 16, Xb + (size_t)(mBase + r) * DMODEL + k0 + kc);
            async_ld_b128(boff + q1 * 16, W  + (size_t)(nBase + r) * DMODEL + k0 + kc);
        }
    };

    v8f acc[2][4] = {};
    issue(0, 0);
    const int KT = DMODEL / 32;          // 16 K-panels
    for (int kt = 0; kt < KT; ++kt) {
        const int cur = kt & 1;
        if (kt + 1 < KT) { issue(cur ^ 1, (kt + 1) * 32); wait_async_le4(); }
        else             { wait_async_0(); }
        __syncthreads();                 // cur panel resident for all waves

        const bf16_t* As = smem + cur * 8192;
        const bf16_t* Bs = As + 4096;
        v16bf a0 = load_a_frag(As, 32, wm * 32 +  0, 0, lane);
        v16bf a1 = load_a_frag(As, 32, wm * 32 + 16, 0, lane);
#pragma unroll
        for (int j = 0; j < 4; ++j) {
            v16bf bf = load_bT_frag(Bs, 32, wn * 64 + 16 * j, 0, lane);
            acc[0][j] = wmma_bf16(a0, bf, acc[0][j]);
            acc[1][j] = wmma_bf16(a1, bf, acc[1][j]);
        }
        __syncthreads();                 // all waves done before cur is refilled
    }
#pragma unroll
    for (int i = 0; i < 2; ++i)
#pragma unroll
        for (int j = 0; j < 4; ++j)
            store_c(C, 2 * DINNER, mBase + wm * 32 + 16 * i,
                    nBase + wn * 64 + 16 * j, lane, acc[i][j]);
}

// ---------------------------------------------------------------------------
// Depthwise causal conv (k=4) + SiLU. Bwd direction = anti-causal on original
// time order (equivalent to reverse -> causal conv -> reverse).
// grid(TOK*DINNER/256, 2)
// ---------------------------------------------------------------------------
__global__ void k_conv_silu(const float* __restrict__ Pf,
                            const float* __restrict__ Pb,
                            const float* __restrict__ cwf, const float* __restrict__ cbf,
                            const float* __restrict__ cwb, const float* __restrict__ cbb,
                            float* __restrict__ Uf32f, bf16_t* __restrict__ Ubff,
                            float* __restrict__ Uf32b, bf16_t* __restrict__ Ubfb) {
    const int idx = blockIdx.x * 256 + threadIdx.x;   // token*DINNER + c
    const int dir = blockIdx.y;
    const int c     = idx & (DINNER - 1);
    const int token = idx >> 10;                       // DINNER = 1024
    const int t     = token & (SEQ - 1);
    const int base  = token - t;                       // b * SEQ
    const float* P  = dir ? Pb  : Pf;
    const float* cw = dir ? cwb : cwf;
    const float* cb = dir ? cbb : cbf;

    float acc = cb[c];
#pragma unroll
    for (int k = 0; k < 4; ++k) {
        const int tt = dir ? (t + 3 - k) : (t - 3 + k);
        if (tt >= 0 && tt < SEQ)
            acc += cw[c * 4 + k] * P[(size_t)(base + tt) * (2 * DINNER) + c];
    }
    const float s = acc / (1.f + __expf(-acc));        // SiLU
    (dir ? Uf32b : Uf32f)[idx] = s;
    (dir ? Ubfb  : Ubff )[idx] = (bf16_t)s;
}

// ---------------------------------------------------------------------------
// dbl[8192,64] = uc @ xp_w^T.  block 256 (8 waves, each a 16-row tile x all 64
// cols). grid(64, 2)
// ---------------------------------------------------------------------------
__global__ void k_gemm_dbl(const bf16_t* __restrict__ Uf, const bf16_t* __restrict__ Ub,
                           const bf16_t* __restrict__ Wxf, const bf16_t* __restrict__ Wxb,
                           float* __restrict__ Dblf, float* __restrict__ Dblb) {
    const int lane = threadIdx.x & 31;
    const int wave = threadIdx.x >> 5;
    const int m0   = (blockIdx.x * 8 + wave) * 16;
    const int dir  = blockIdx.y;
    const bf16_t* A = dir ? Ub  : Uf;
    const bf16_t* W = dir ? Wxb : Wxf;        // [64, 1024]
    float*        C = dir ? Dblb : Dblf;

    v8f a0 = {}, a1 = {}, a2 = {}, a3 = {};
    for (int k0 = 0; k0 < DINNER; k0 += 32) {
        v16bf a = load_a_frag(A, DINNER, m0, k0, lane);
        a0 = wmma_bf16(a, load_bT_frag(W, DINNER,  0, k0, lane), a0);
        a1 = wmma_bf16(a, load_bT_frag(W, DINNER, 16, k0, lane), a1);
        a2 = wmma_bf16(a, load_bT_frag(W, DINNER, 32, k0, lane), a2);
        a3 = wmma_bf16(a, load_bT_frag(W, DINNER, 48, k0, lane), a3);
    }
    store_c(C, 64, m0,  0, lane, a0);
    store_c(C, 64, m0, 16, lane, a1);
    store_c(C, 64, m0, 32, lane, a2);
    store_c(C, 64, m0, 48, lane, a3);
}

// ---------------------------------------------------------------------------
// delta[8192,1024] = softplus(dbl[:, :32] @ dt_w^T + dt_b).  K == 32: exactly
// one WMMA per 16x16 tile; softplus fused in epilogue. grid(512, 2, 2)
// ---------------------------------------------------------------------------
__global__ void k_gemm_delta(const float* __restrict__ Dblf, const float* __restrict__ Dblb,
                             const bf16_t* __restrict__ Wdf, const bf16_t* __restrict__ Wdb,
                             const float* __restrict__ dtbf, const float* __restrict__ dtbb,
                             float* __restrict__ Delf, float* __restrict__ Delb) {
    const int lane = threadIdx.x & 31;
    const int wave = threadIdx.x >> 5;
    const int m0   = blockIdx.x * 16;
    const int n0   = (blockIdx.y * 8 + wave) * 64;
    const int dir  = blockIdx.z;
    const float*  Dbl  = dir ? Dblb : Dblf;
    const bf16_t* W    = dir ? Wdb  : Wdf;    // [1024, 32]
    const float*  bias = dir ? dtbb : dtbf;
    float*        Out  = dir ? Delb : Delf;

    // Build A fragment from f32 dbl columns [0,32), converting to bf16.
    const int half = lane >> 4;
    const int m    = m0 + (lane & 15);
    const float* ap = Dbl + (size_t)m * 64 + half * 16;
    v16bf a;
#pragma unroll
    for (int i = 0; i < 8; ++i) { a[i] = (bf16_t)ap[i]; a[8 + i] = (bf16_t)ap[8 + i]; }

#pragma unroll
    for (int j = 0; j < 4; ++j) {
        v8f acc = {};
        acc = wmma_bf16(a, load_bT_frag(W, DTRANK, n0 + 16 * j, 0, lane), acc);
        const int   n  = n0 + 16 * j + (lane & 15);
        const float bv = bias[n];
#pragma unroll
        for (int r = 0; r < 8; ++r) {
            const float v  = acc[r] + bv;
            const float sp = (v > 20.f) ? v : log1pf(__expf(v));   // softplus
            Out[(size_t)(m0 + r + 8 * half) * DINNER + n] = sp;
        }
    }
}

// ---------------------------------------------------------------------------
// Selective scan + gating.  One thread per (channel, state): state h lives in
// a register for all 1024 steps.  16-lane xor-reduction (half of a wave32)
// forms y = sum_s h*C.  Output g = (y + u*D) * silu(z) stored as bf16 for the
// final WMMA GEMM.  grid(64, 8, 2), block 256 (16 channels per block).
// ---------------------------------------------------------------------------
__global__ void k_scan_gate(const float* __restrict__ Uf32f, const float* __restrict__ Uf32b,
                            const float* __restrict__ Dblf,  const float* __restrict__ Dblb,
                            const float* __restrict__ Delf,  const float* __restrict__ Delb,
                            const float* __restrict__ Alogf, const float* __restrict__ Alogb,
                            const float* __restrict__ Dpf,   const float* __restrict__ Dpb,
                            const float* __restrict__ Pf,    const float* __restrict__ Pb,
                            bf16_t* __restrict__ Gf, bf16_t* __restrict__ Gb) {
    const int s   = threadIdx.x & 15;          // state index
    const int c   = blockIdx.x * 16 + (threadIdx.x >> 4);
    const int b   = blockIdx.y;
    const int dir = blockIdx.z;
    const float* uc    = dir ? Uf32b : Uf32f;
    const float* dbl   = dir ? Dblb  : Dblf;
    const float* delta = dir ? Delb  : Delf;
    const float* Alog  = dir ? Alogb : Alogf;
    const float* Dp    = dir ? Dpb   : Dpf;
    const float* P     = dir ? Pb    : Pf;
    bf16_t*      G     = dir ? Gb    : Gf;

    const float A  = -__expf(Alog[c * DSTATE + s]);
    const float Dc = Dp[c];
    float h = 0.f;

    for (int step = 0; step < SEQ; ++step) {
        const int    t     = dir ? (SEQ - 1 - step) : step;
        const size_t token = (size_t)b * SEQ + t;

        const float d  = delta[token * DINNER + c];
        const float u  = uc[token * DINNER + c];
        const float Bt = dbl[token * 64 + DTRANK + s];
        const float Ct = dbl[token * 64 + DTRANK + DSTATE + s];

        h = h * __expf(d * A) + (d * u) * Bt;
        float y = h * Ct;
        y += __shfl_xor(y, 1, 32);
        y += __shfl_xor(y, 2, 32);
        y += __shfl_xor(y, 4, 32);
        y += __shfl_xor(y, 8, 32);

        if (s == 0) {
            const float z = P[token * (2 * DINNER) + DINNER + c];
            const float g = (y + u * Dc) * (z / (1.f + __expf(-z)));
            G[token * DINNER + c] = (bf16_t)g;
        }

        // prefetch next step's operands (global_prefetch_b8)
        const int tn = dir ? (t - 1) : (t + 1);
        if (tn >= 0 && tn < SEQ) {
            const size_t tok2 = (size_t)b * SEQ + tn;
            __builtin_prefetch(&delta[tok2 * DINNER + c], 0, 1);
            __builtin_prefetch(&dbl[tok2 * 64 + DTRANK + s], 0, 1);
        }
    }
}

// ---------------------------------------------------------------------------
// out[8192,512] = Gf @ out_w_f^T + Gb @ out_w_b^T.  One block covers all 512
// output cols of one 16-row tile (8 waves x 64 cols). grid(512), block 256.
// ---------------------------------------------------------------------------
__global__ void k_gemm_out(const bf16_t* __restrict__ Gf, const bf16_t* __restrict__ Gb,
                           const bf16_t* __restrict__ Wof, const bf16_t* __restrict__ Wob,
                           float* __restrict__ Out) {
    const int lane = threadIdx.x & 31;
    const int wave = threadIdx.x >> 5;
    const int m0   = blockIdx.x * 16;
    const int n0   = wave * 64;

    v8f a0 = {}, a1 = {}, a2 = {}, a3 = {};
    for (int k0 = 0; k0 < DINNER; k0 += 32) {
        v16bf a = load_a_frag(Gf, DINNER, m0, k0, lane);
        a0 = wmma_bf16(a, load_bT_frag(Wof, DINNER, n0 +  0, k0, lane), a0);
        a1 = wmma_bf16(a, load_bT_frag(Wof, DINNER, n0 + 16, k0, lane), a1);
        a2 = wmma_bf16(a, load_bT_frag(Wof, DINNER, n0 + 32, k0, lane), a2);
        a3 = wmma_bf16(a, load_bT_frag(Wof, DINNER, n0 + 48, k0, lane), a3);
    }
    for (int k0 = 0; k0 < DINNER; k0 += 32) {
        v16bf a = load_a_frag(Gb, DINNER, m0, k0, lane);
        a0 = wmma_bf16(a, load_bT_frag(Wob, DINNER, n0 +  0, k0, lane), a0);
        a1 = wmma_bf16(a, load_bT_frag(Wob, DINNER, n0 + 16, k0, lane), a1);
        a2 = wmma_bf16(a, load_bT_frag(Wob, DINNER, n0 + 32, k0, lane), a2);
        a3 = wmma_bf16(a, load_bT_frag(Wob, DINNER, n0 + 48, k0, lane), a3);
    }
    store_c(Out, DMODEL, m0, n0 +  0, lane, a0);
    store_c(Out, DMODEL, m0, n0 + 16, lane, a1);
    store_c(Out, DMODEL, m0, n0 + 32, lane, a2);
    store_c(Out, DMODEL, m0, n0 + 48, lane, a3);
}

// ---------------------------------------------------------------------------
// Host launch
// ---------------------------------------------------------------------------
static inline size_t align256(size_t x) { return (x + 255) & ~(size_t)255; }

extern "C" void kernel_launch(void* const* d_in, const int* in_sizes, int n_in,
                              void* d_out, int out_size, void* d_ws, size_t ws_size,
                              hipStream_t stream) {
    // inputs (setup_inputs order)
    const float* x      = (const float*)d_in[0];
    const float* in_w_f = (const float*)d_in[1];
    const float* cw_f   = (const float*)d_in[2];
    const float* cb_f   = (const float*)d_in[3];
    const float* xp_w_f = (const float*)d_in[4];
    const float* dt_w_f = (const float*)d_in[5];
    const float* dt_b_f = (const float*)d_in[6];
    const float* Alog_f = (const float*)d_in[7];
    const float* Dp_f   = (const float*)d_in[8];
    const float* out_w_f= (const float*)d_in[9];
    const float* in_w_b = (const float*)d_in[10];
    const float* cw_b   = (const float*)d_in[11];
    const float* cb_b   = (const float*)d_in[12];
    const float* xp_w_b = (const float*)d_in[13];
    const float* dt_w_b = (const float*)d_in[14];
    const float* dt_b_b = (const float*)d_in[15];
    const float* Alog_b = (const float*)d_in[16];
    const float* Dp_b   = (const float*)d_in[17];
    const float* out_w_b= (const float*)d_in[18];
    float* out = (float*)d_out;

    // workspace bump allocator (deterministic)
    char*  ws  = (char*)d_ws;
    size_t off = 0;
    auto alloc = [&](size_t bytes) { void* p = ws + off; off = align256(off + bytes); return p; };

    bf16_t* xb   = (bf16_t*)alloc((size_t)TOK * DMODEL * 2);
    bf16_t* wi_f = (bf16_t*)alloc((size_t)2 * DINNER * DMODEL * 2);
    bf16_t* wi_b = (bf16_t*)alloc((size_t)2 * DINNER * DMODEL * 2);
    bf16_t* wx_f = (bf16_t*)alloc((size_t)64 * DINNER * 2);
    bf16_t* wx_b = (bf16_t*)alloc((size_t)64 * DINNER * 2);
    bf16_t* wd_f = (bf16_t*)alloc((size_t)DINNER * DTRANK * 2);
    bf16_t* wd_b = (bf16_t*)alloc((size_t)DINNER * DTRANK * 2);
    bf16_t* wo_f = (bf16_t*)alloc((size_t)DMODEL * DINNER * 2);
    bf16_t* wo_b = (bf16_t*)alloc((size_t)DMODEL * DINNER * 2);
    float*  Pf   = (float*)alloc((size_t)TOK * 2 * DINNER * 4);
    float*  Pb   = (float*)alloc((size_t)TOK * 2 * DINNER * 4);
    float*  Uf32f= (float*)alloc((size_t)TOK * DINNER * 4);
    float*  Uf32b= (float*)alloc((size_t)TOK * DINNER * 4);
    bf16_t* Ubff = (bf16_t*)alloc((size_t)TOK * DINNER * 2);
    bf16_t* Ubfb = (bf16_t*)alloc((size_t)TOK * DINNER * 2);
    float*  Dblf = (float*)alloc((size_t)TOK * 64 * 4);
    float*  Dblb = (float*)alloc((size_t)TOK * 64 * 4);
    float*  Delf = (float*)alloc((size_t)TOK * DINNER * 4);
    float*  Delb = (float*)alloc((size_t)TOK * DINNER * 4);
    bf16_t* Gf   = (bf16_t*)alloc((size_t)TOK * DINNER * 2);
    bf16_t* Gb   = (bf16_t*)alloc((size_t)TOK * DINNER * 2);
    (void)ws_size; (void)n_in; (void)in_sizes; (void)out_size;

    auto cvt = [&](const float* s, bf16_t* d, int n) {
        k_cvt_bf16<<<dim3((n + 255) / 256), dim3(256), 0, stream>>>(s, d, n);
    };
    cvt(x,       xb,   TOK * DMODEL);
    cvt(in_w_f,  wi_f, 2 * DINNER * DMODEL);
    cvt(in_w_b,  wi_b, 2 * DINNER * DMODEL);
    cvt(xp_w_f,  wx_f, 64 * DINNER);
    cvt(xp_w_b,  wx_b, 64 * DINNER);
    cvt(dt_w_f,  wd_f, DINNER * DTRANK);
    cvt(dt_w_b,  wd_b, DINNER * DTRANK);
    cvt(out_w_f, wo_f, DMODEL * DINNER);
    cvt(out_w_b, wo_b, DMODEL * DINNER);

    // staged WMMA GEMM: 128x128 tiles, async double-buffered K panels
    k_gemm_inproj<<<dim3(TOK / 128, (2 * DINNER) / 128, 2), dim3(256), 0, stream>>>(
        xb, wi_f, wi_b, Pf, Pb);

    k_conv_silu<<<dim3((TOK * DINNER) / 256, 2), dim3(256), 0, stream>>>(
        Pf, Pb, cw_f, cb_f, cw_b, cb_b, Uf32f, Ubff, Uf32b, Ubfb);

    k_gemm_dbl<<<dim3(TOK / 128, 2), dim3(256), 0, stream>>>(Ubff, Ubfb, wx_f, wx_b, Dblf, Dblb);

    k_gemm_delta<<<dim3(TOK / 16, 2, 2), dim3(256), 0, stream>>>(
        Dblf, Dblb, wd_f, wd_b, dt_b_f, dt_b_b, Delf, Delb);

    k_scan_gate<<<dim3(DINNER / 16, BATCH, 2), dim3(256), 0, stream>>>(
        Uf32f, Uf32b, Dblf, Dblb, Delf, Delb, Alog_f, Alog_b, Dp_f, Dp_b, Pf, Pb, Gf, Gb);

    k_gemm_out<<<dim3(TOK / 16), dim3(256), 0, stream>>>(Gf, Gb, wo_f, wo_b, out);
}
